// RNN_MultiRegional_D1D2_58420145160253
// MI455X (gfx1250) — matrix operand order repaired and checked by reading, not compiled, not run on hardware
//
#include <hip/hip_runtime.h>

#define HID   256
#define INPN  64
#define FSIN  76
#define TOT   1676
#define BATCH 32
#define STEPS 1000
#define H2    128
#define ALM_CUT 180                 // HID - INH (INH = 76)
#define KT_TILES 53                 // ceil(1676/32)
#define KP  (KT_TILES * 32)         // 1696 padded K
#define NT_TILES 105                // ceil(1676/16)
#define TCONST 0.01f
#define NSCALE 0.0014142135623730951f   // sqrt(2*0.01)*0.01

typedef __attribute__((ext_vector_type(16))) __bf16 v16bf;
typedef __attribute__((ext_vector_type(8)))  __bf16 v8bf;
typedef __attribute__((ext_vector_type(8)))  float  v8f;

struct WPtrs {
    const float* sparse_mask;
    const float* str2str;  const float* thal2alm; const float* thal2str;
    const float* alm2alm;  const float* alm2str;  const float* alm2thal;
    const float* str2snr;  const float* str2gpe;  const float* gpe2stn;
    const float* stn2snr;  const float* snr2thal; const float* fsi2str;
    const float* thal2fsi; const float* alm2fsi;  const float* iti2fsi;
    const float* fsi2fsi;  const float* inp_str;
};

__device__ __forceinline__ float relu_(float v) { return v > 0.f ? v : 0.f; }

// W[j][k] of the assembled 1676x1676 connectivity matrix (row j = target, col k = source)
__device__ float compute_W(int j, int k, const WPtrs& p) {
    if (j < HID) {                                  // striatum rows
        if (k < HID) {
            if ((j < H2) != (k < H2)) return 0.f;   // SPLIT block diag
            return -(p.sparse_mask[j * HID + k] * relu_(p.str2str[j * HID + k]));
        }
        if (k < 4 * HID) return 0.f;
        if (k < 5 * HID) return relu_(p.thal2str[j * HID + (k - 4 * HID)]);
        if (k < 6 * HID) { int c = k - 5 * HID; return (c >= ALM_CUT) ? 0.f : relu_(p.alm2str[j * HID + c]); }
        if (k < 6 * HID + INPN) return relu_(p.inp_str[j * INPN + (k - 6 * HID)]);
        return -relu_(p.fsi2str[j * FSIN + (k - (6 * HID + INPN))]);
    }
    if (j < 2 * HID) {                              // GPe rows
        int r = j - HID;
        if (k < HID) return (k >= H2) ? -relu_(p.str2gpe[r * HID + k]) : 0.f;
        return 0.f;
    }
    if (j < 3 * HID) {                              // STN rows
        int r = j - 2 * HID;
        if (k >= HID && k < 2 * HID) return -relu_(p.gpe2stn[r * HID + (k - HID)]);
        return 0.f;
    }
    if (j < 4 * HID) {                              // SNr rows
        int r = j - 3 * HID;
        if (k < HID) return (k < H2) ? -relu_(p.str2snr[r * HID + k]) : 0.f;
        if (k >= 2 * HID && k < 3 * HID) return relu_(p.stn2snr[r * HID + (k - 2 * HID)]);
        return 0.f;
    }
    if (j < 5 * HID) {                              // thalamus rows
        int r = j - 4 * HID;
        if (k >= 3 * HID && k < 4 * HID) return -relu_(p.snr2thal[r * HID + (k - 3 * HID)]);
        if (k >= 5 * HID && k < 6 * HID) { int c = k - 5 * HID; return (c >= ALM_CUT) ? 0.f : relu_(p.alm2thal[r * HID + c]); }
        return 0.f;
    }
    if (j < 6 * HID) {                              // ALM rows
        int r = j - 5 * HID;
        if (k >= 4 * HID && k < 5 * HID) return relu_(p.thal2alm[r * HID + (k - 4 * HID)]);
        if (k >= 5 * HID && k < 6 * HID) {
            int c = k - 5 * HID;
            float s = (c >= ALM_CUT) ? -1.f : 1.f;  // Dale diagonal
            return s * relu_(p.alm2alm[r * HID + c]);
        }
        return 0.f;
    }
    if (j < 6 * HID + INPN) return 0.f;             // ITI rows: zero
    {                                               // FSI rows
        int r = j - (6 * HID + INPN);
        if (k >= 4 * HID && k < 5 * HID) return relu_(p.thal2fsi[r * HID + (k - 4 * HID)]);
        if (k >= 5 * HID && k < 6 * HID) return relu_(p.alm2fsi[r * HID + (k - 5 * HID)]);
        if (k >= 6 * HID && k < 6 * HID + INPN) return relu_(p.iti2fsi[r * INPN + (k - 6 * HID)]);
        if (k >= 6 * HID + INPN && k < TOT) return -relu_(p.fsi2fsi[r * FSIN + (k - (6 * HID + INPN))]);
        return 0.f;
    }
}

// Build W^T in a WMMA-B-fragment blocked bf16 layout:
// wblk[((kt*NT + nt)*32 + lane)*16 + e] = W[j = nt*16 + (lane&15)][k = kt*32 + (e>>3)*16 + (lane>>4)*8 + (e&7)]
__global__ void build_w_kernel(WPtrs p, __bf16* __restrict__ wblk) {
    int idx = blockIdx.x * 256 + threadIdx.x;
    const int total = KT_TILES * NT_TILES * 512;
    if (idx >= total) return;
    int e = idx & 15;
    int lane = (idx >> 4) & 31;
    int tile = idx >> 9;
    int nt = tile % NT_TILES;
    int kt = tile / NT_TILES;
    int j = nt * 16 + (lane & 15);
    int k = kt * 32 + (e >> 3) * 16 + (lane >> 4) * 8 + (e & 7);
    float w = (j < TOT && k < TOT) ? compute_W(j, k, p) : 0.f;
    wblk[idx] = (__bf16)w;
}

__global__ void init_kernel(const float* __restrict__ hn, const float* __restrict__ xn,
                            __bf16* __restrict__ h0, __bf16* __restrict__ h1,
                            float* __restrict__ xv, unsigned* __restrict__ bar) {
    int idx = blockIdx.x * 256 + threadIdx.x;
    if (idx == 0) { bar[0] = 0u; bar[1] = 0u; }
    if (idx >= BATCH * KP) return;
    int b = idx / KP, j = idx % KP;
    float hv = (j < TOT) ? hn[b * TOT + j] : 0.f;
    __bf16 hb = (__bf16)hv;
    h0[idx] = hb;
    h1[idx] = hb;
    if (j < TOT) xv[b * TOT + j] = xn[b * TOT + j];
}

// Persistent RNN: 27 blocks x 256 threads (8 waves). wave -> (ntile, mtile).
// Double-buffered bf16 h, device-wide generation barrier between steps.
__global__ void __launch_bounds__(256, 1) rnn_persistent(
    const float* __restrict__ inp, const float* __restrict__ cue,
    const float* __restrict__ inhib, const float* __restrict__ noise,
    const __bf16* __restrict__ wblk, __bf16* __restrict__ h0, __bf16* __restrict__ h1,
    float* __restrict__ xv, float* __restrict__ out,
    unsigned* __restrict__ bar_cnt, unsigned* __restrict__ bar_gen, int nblocks)
{
    const int tid  = threadIdx.x;
    const int wave = tid >> 5;
    const int lane = tid & 31;
    const int nt   = blockIdx.x * 4 + (wave >> 1);  // output-neuron tile
    const int mt   = wave & 1;                      // batch tile (0: b 0-15, 1: b 16-31)
    const int half = lane >> 4;
    const int n    = lane & 15;
    const int j    = nt * 16 + n;
    const bool active = (nt < NT_TILES);            // uniform per wave
    const int arow = mt * 16 + n;                   // A-matrix row = batch index (lane&15)

    // static per-neuron drive masks
    float tonic = 0.f, itim = 0.f, thalm = 0.f;
    if (j >= 1 * HID && j < 3 * HID) tonic = 1.f;               // gpe, stn
    if (j >= 4 * HID && j < 5 * HID) { tonic = 1.f; thalm = 1.f; } // thal
    if (j >= 6 * HID && j < 6 * HID + INPN) itim = 1.f;         // iti
    const bool jval = active && (j < TOT);

    for (int t = 0; t < STEPS; ++t) {
        const __bf16* __restrict__ Hr = (t & 1) ? h1 : h0;
        __bf16* __restrict__ Hw       = (t & 1) ? h0 : h1;

        if (active) {
            v8f acc = {0.f, 0.f, 0.f, 0.f, 0.f, 0.f, 0.f, 0.f};
            const __bf16* ap = Hr + (size_t)arow * KP + half * 8;
            const __bf16* bp = wblk + ((size_t)nt << 9) + (lane << 4);
            for (int kt = 0; kt < KT_TILES; ++kt) {
                // A frag (16x32 bf16): lane half selects K 0-7/16-23 vs 8-15/24-31
                v8bf a0 = *(const v8bf*)(ap);
                v8bf a1 = *(const v8bf*)(ap + 16);
                // B frag (32x16 bf16): pre-blocked, 32B contiguous per lane
                v8bf b0 = *(const v8bf*)(bp);
                v8bf b1 = *(const v8bf*)(bp + 8);
                v16bf av = __builtin_shufflevector(a0, a1, 0,1,2,3,4,5,6,7,8,9,10,11,12,13,14,15);
                v16bf bv = __builtin_shufflevector(b0, b1, 0,1,2,3,4,5,6,7,8,9,10,11,12,13,14,15);
                acc = __builtin_amdgcn_wmma_f32_16x16x32_bf16(
                        false, av, false, bv, (short)0, acc, false, false);
                ap += 32;
                bp += (size_t)NT_TILES << 9;
            }
            if (jval) {
                #pragma unroll
                for (int r = 0; r < 8; ++r) {
                    int b = mt * 16 + half * 8 + r;     // C/D layout: VGPR r -> M=r(+8)
                    float inp_t = inp[b * STEPS + t];
                    float cue_t = cue[b * STEPS + t];
                    size_t o = ((size_t)b * STEPS + t) * TOT + j;
                    float drive = acc[r] + tonic + inp_t * itim + cue_t * thalm
                                + inhib[o] + NSCALE * noise[o];
                    float x = xv[b * TOT + j];
                    x += TCONST * (drive - x);
                    xv[b * TOT + j] = x;
                    float h = fmaxf(x, 0.f);
                    out[o] = h;
                    Hw[(size_t)b * KP + j] = (__bf16)h;
                }
            }
        }

        // ---- device-wide barrier (generation counter) ----
        __threadfence();
        __syncthreads();
        if (tid == 0) {
            unsigned target = (unsigned)(t + 1);
            unsigned prev = __hip_atomic_fetch_add(bar_cnt, 1u, __ATOMIC_ACQ_REL,
                                                   __HIP_MEMORY_SCOPE_AGENT);
            if (prev == (unsigned)nblocks - 1u) {
                __hip_atomic_store(bar_cnt, 0u, __ATOMIC_RELAXED, __HIP_MEMORY_SCOPE_AGENT);
                __hip_atomic_store(bar_gen, target, __ATOMIC_RELEASE, __HIP_MEMORY_SCOPE_AGENT);
            } else {
                while (__hip_atomic_load(bar_gen, __ATOMIC_ACQUIRE,
                                         __HIP_MEMORY_SCOPE_AGENT) < target) {
                    __builtin_amdgcn_s_sleep(2);
                }
            }
        }
        __syncthreads();
        __threadfence();
    }
}

extern "C" void kernel_launch(void* const* d_in, const int* in_sizes, int n_in,
                              void* d_out, int out_size, void* d_ws, size_t ws_size,
                              hipStream_t stream) {
    const float* inp   = (const float*)d_in[0];
    const float* cue   = (const float*)d_in[1];
    const float* hn    = (const float*)d_in[2];
    const float* xn    = (const float*)d_in[3];
    const float* inhib = (const float*)d_in[4];
    const float* noise = (const float*)d_in[5];

    WPtrs p;
    p.sparse_mask = (const float*)d_in[6];
    p.str2str  = (const float*)d_in[7];
    p.thal2alm = (const float*)d_in[8];
    p.thal2str = (const float*)d_in[9];
    p.alm2alm  = (const float*)d_in[10];
    p.alm2str  = (const float*)d_in[11];
    p.alm2thal = (const float*)d_in[12];
    p.str2snr  = (const float*)d_in[13];
    p.str2gpe  = (const float*)d_in[14];
    p.gpe2stn  = (const float*)d_in[15];
    p.stn2snr  = (const float*)d_in[16];
    p.snr2thal = (const float*)d_in[17];
    p.fsi2str  = (const float*)d_in[18];
    p.thal2fsi = (const float*)d_in[19];
    p.alm2fsi  = (const float*)d_in[20];
    p.iti2fsi  = (const float*)d_in[21];
    p.fsi2fsi  = (const float*)d_in[22];
    p.inp_str  = (const float*)d_in[23];

    // workspace layout (all 256B aligned)
    const size_t WBLK_BYTES = (size_t)KT_TILES * NT_TILES * 512 * sizeof(__bf16); // 5,698,560
    const size_t HBUF_BYTES = (size_t)BATCH * KP * sizeof(__bf16);                // 108,544
    char* ws = (char*)d_ws;
    __bf16* wblk = (__bf16*)ws;
    __bf16* hb0  = (__bf16*)(ws + WBLK_BYTES);
    __bf16* hb1  = (__bf16*)(ws + WBLK_BYTES + HBUF_BYTES);
    float*  xv   = (float*)(ws + WBLK_BYTES + 2 * HBUF_BYTES);
    unsigned* bar = (unsigned*)(ws + WBLK_BYTES + 2 * HBUF_BYTES + (size_t)BATCH * TOT * sizeof(float));

    // 1) reset state (h ping-pong, xv, barrier) — re-done every call: deterministic
    {
        int n = BATCH * KP;
        init_kernel<<<(n + 255) / 256, 256, 0, stream>>>(hn, xn, hb0, hb1, xv, bar);
    }
    // 2) build blocked bf16 W^T
    {
        int n = KT_TILES * NT_TILES * 512;
        build_w_kernel<<<(n + 255) / 256, 256, 0, stream>>>(p, wblk);
    }
    // 3) one persistent kernel for all 1000 steps
    const int nblocks = (NT_TILES + 3) / 4;  // 27 blocks x 8 waves, 4 n-tiles each
    rnn_persistent<<<nblocks, 256, 0, stream>>>(inp, cue, inhib, noise, wblk,
                                                hb0, hb1, xv, (float*)d_out,
                                                bar, bar + 1, nblocks);
}
